// MultiHeadFAVORAttention_63187558859248
// MI455X (gfx1250) — matrix operand order repaired and checked by reading, not compiled, not run on hardware
//
#include <hip/hip_runtime.h>
#include <hip/hip_bf16.h>

// ---------------------------------------------------------------------------
// FAVOR+ causal attention for MI455X (gfx1250), chunkwise linear-attention
// formulation so every matrix product maps onto v_wmma_f32_16x16x32_bf16.
// ---------------------------------------------------------------------------

#define HEADS 8
#define DIMM  512
#define DK    64
#define MFEAT 256
#define BB    4
#define LL    2048
#define NROWS (BB * LL)
#define STABV 1e-5f

typedef __bf16 bf16_t;
typedef __attribute__((ext_vector_type(16))) __bf16 v16bf;
typedef __attribute__((ext_vector_type(8)))  __bf16 bf16x8;
typedef __attribute__((ext_vector_type(8)))  float  v8f;

#define WMMA_BF16(a, b, c) \
    __builtin_amdgcn_wmma_f32_16x16x32_bf16(false, (a), false, (b), (short)0, (c), false, false)

// A-matrix fragment (16x32 bf16), ISA layout:
// lane: row = lane&15 ; lane<16: e0..7->K0..7, e8..15->K16..23
//                       lane>=16: e0..7->K8..15, e8..15->K24..31
static __device__ inline v16bf frag_a_lds(const bf16_t* base, int stride) {
    int lane = threadIdx.x & 31;
    int row = lane & 15, hi = lane >> 4;
    const bf16_t* p = base + row * stride + hi * 8;
    bf16x8 lo = *(const bf16x8*)(p);
    bf16x8 hp = *(const bf16x8*)(p + 16);
    return __builtin_shufflevector(lo, hp, 0,1,2,3,4,5,6,7,8,9,10,11,12,13,14,15);
}

// B-matrix fragment (32x16 bf16, K x N), stored in LDS as [n][k]:
// lane: n = lane&15 ; lane<16: e0..7->K0..7, e8..15->K16..23
//                     lane>=16: e0..7->K8..15, e8..15->K24..31
static __device__ inline v16bf frag_b_lds(const bf16_t* base, int stride) {
    int lane = threadIdx.x & 31;
    int n = lane & 15, kb = (lane >> 4) * 8;
    const bf16_t* p = base + n * stride + kb;
    bf16x8 lo = *(const bf16x8*)(p);
    bf16x8 hp = *(const bf16x8*)(p + 16);
    return __builtin_shufflevector(lo, hp, 0,1,2,3,4,5,6,7,8,9,10,11,12,13,14,15);
}

// ---------------------------------------------------------------------------
// Generic GEMM: C[r, c] = act( scale * sum_k A[r, k0+k] * W[c, k] + bias[c] )
// A fp32 row-major, W fp32 row-major [cols, K] (i.e. multiplied transposed).
// Output either fp32 or bf16, with row stride / per-z column offset (heads).
// Block tile 128x64, 8 waves (4x2), each wave 2x2 WMMA tiles, K-step 32.
// ---------------------------------------------------------------------------
#define BM 128
#define BN 64
#define BK 32
#define LDT (BK + 8)

__global__ __launch_bounds__(256)
void gemm_nt(const float* __restrict__ A, int lda, int a_col_z,
             const float* __restrict__ W, int ldw,
             const float* __restrict__ bias,
             float scale, int relu_stab,
             float* __restrict__ outF, bf16_t* __restrict__ outB,
             long out_rstride, long out_col_z, int K)
{
    __shared__ __align__(16) bf16_t sA[BM][LDT];
    __shared__ __align__(16) bf16_t sB[BN][LDT];

    const int tid  = threadIdx.x;
    const int lane = tid & 31;
    const int wave = tid >> 5;
    const int row0 = blockIdx.y * BM;
    const int col0 = blockIdx.x * BN;
    const int acol = blockIdx.z * a_col_z;
    const long ocol0 = (long)blockIdx.z * out_col_z + col0;

    const int wr = wave >> 1, wc = wave & 1;     // waves arranged 4 x 2
    v8f acc[2][2] = {};

    const int arow = tid >> 1, aseg = (tid & 1) * 16;   // A: 16 floats / thread
    const int brow = tid >> 2, bseg = (tid & 3) * 8;    // B: 8 floats / thread

    for (int kt = 0; kt < K; kt += BK) {
        const float* ga = A + (long)(row0 + arow) * lda + acol + kt + aseg;
        float4 a0 = ((const float4*)ga)[0];
        float4 a1 = ((const float4*)ga)[1];
        float4 a2 = ((const float4*)ga)[2];
        float4 a3 = ((const float4*)ga)[3];
        const float* gb = W + (long)(col0 + brow) * ldw + kt + bseg;
        float4 b0 = ((const float4*)gb)[0];
        float4 b1 = ((const float4*)gb)[1];

        if (kt + BK < K) {                       // gfx1250 global_prefetch_b8
            __builtin_prefetch(ga + BK, 0, 0);
            __builtin_prefetch(gb + BK, 0, 0);
        }

        {
            float av[16] = { a0.x,a0.y,a0.z,a0.w, a1.x,a1.y,a1.z,a1.w,
                             a2.x,a2.y,a2.z,a2.w, a3.x,a3.y,a3.z,a3.w };
            bf16_t* da = &sA[arow][aseg];
            #pragma unroll
            for (int e = 0; e < 16; ++e) da[e] = (bf16_t)av[e];
            float bvv[8] = { b0.x,b0.y,b0.z,b0.w, b1.x,b1.y,b1.z,b1.w };
            bf16_t* db = &sB[brow][bseg];
            #pragma unroll
            for (int e = 0; e < 8; ++e) db[e] = (bf16_t)bvv[e];
        }
        __syncthreads();

        #pragma unroll
        for (int i = 0; i < 2; ++i) {
            v16bf af = frag_a_lds(&sA[(wr * 2 + i) * 16][0], LDT);
            #pragma unroll
            for (int j = 0; j < 2; ++j) {
                v16bf bfv = frag_b_lds(&sB[(wc * 2 + j) * 16][0], LDT);
                acc[i][j] = WMMA_BF16(af, bfv, acc[i][j]);
            }
        }
        __syncthreads();
    }

    // epilogue: C/D layout -> lane holds column n, rows r + 8*hi
    const int n = lane & 15, hi = lane >> 4;
    #pragma unroll
    for (int i = 0; i < 2; ++i) {
        #pragma unroll
        for (int j = 0; j < 2; ++j) {
            int crow  = row0 + (wr * 2 + i) * 16 + hi * 8;
            int ccloc = (wc * 2 + j) * 16 + n;
            long ccol = ocol0 + ccloc;
            float bv = bias ? bias[col0 + ccloc] : 0.f;
            #pragma unroll
            for (int r = 0; r < 8; ++r) {
                float v = acc[i][j][r] * scale + bv;
                if (relu_stab) v = fmaxf(v, 0.f) + STABV;
                long idx = (long)(crow + r) * out_rstride + ccol;
                if (outF) outF[idx] = v;
                else      outB[idx] = (bf16_t)v;
            }
        }
    }
}

// ---------------------------------------------------------------------------
// Chunkwise causal linear attention. One workgroup per (b, h); sequential
// over 32 chunks of 64 rows. State S[256][64] fp32 (+ bf16 transposed shadow
// Sbf[64][256] for vectorized B-fragments) + Sk[256] live in LDS (~218 KB of
// the 320 KB WGP LDS; one workgroup per WGP by design).
// Per chunk (all WMMA): A = qp kp^T (masked), num = qp S + A v,
// den = qp.Sk + rowsum(A), out = num/den, S += kp^T v, Sk += colsum(kp).
// ---------------------------------------------------------------------------
#define QPS 264      // bf16 row stride of qp/kp chunk and Sbf (16B-aligned rows)
#define VTS 72       // bf16 row stride of v^T / A_mask / kp^T
#define CHUNK 64

#define OFF_S    0          // float [256][64]   65536 B
#define OFF_SK   65536      // float [256]        1024 B
#define OFF_DEN  66560      // float [64]          256 B
#define OFF_QPC  66816      // bf16 [64][QPS]    33792 B
#define OFF_KPC  100608     // bf16 [64][QPS]    33792 B
#define OFF_VT   134400     // bf16 [64][VTS]     9216 B
#define OFF_AM   143616     // bf16 [64][VTS]     9216 B
#define OFF_SBF  152832     // bf16 [64][QPS]    33792 B  (S^T, bf16)
#define OFF_KPT  186624     // bf16 [256][VTS]   36864 B  (kp^T)
#define SMEM_TOTAL 223488

__global__ __launch_bounds__(256)
void favor_scan(const bf16_t* __restrict__ qp, const bf16_t* __restrict__ kp,
                const float* __restrict__ v, float* __restrict__ attn)
{
    extern __shared__ __align__(16) char smem[];
    float*  S    = (float*)(smem + OFF_S);
    float*  Sk   = (float*)(smem + OFF_SK);
    float*  den  = (float*)(smem + OFF_DEN);
    bf16_t* qpc  = (bf16_t*)(smem + OFF_QPC);
    bf16_t* kpc  = (bf16_t*)(smem + OFF_KPC);
    bf16_t* vT   = (bf16_t*)(smem + OFF_VT);
    bf16_t* Am   = (bf16_t*)(smem + OFF_AM);
    bf16_t* Sbf  = (bf16_t*)(smem + OFF_SBF);
    bf16_t* kpT  = (bf16_t*)(smem + OFF_KPT);

    const int tid  = threadIdx.x;
    const int lane = tid & 31;
    const int wave = tid >> 5;
    const int b = blockIdx.x / HEADS, h = blockIdx.x % HEADS;

    const bf16_t* qpg = qp + (long)b * LL * (HEADS * MFEAT) + h * MFEAT;
    const bf16_t* kpg = kp + (long)b * LL * (HEADS * MFEAT) + h * MFEAT;
    const float*  vg  = v  + (long)b * LL * DIMM + h * DK;
    float*        og  = attn + (long)b * LL * DIMM + h * DK;

    for (int i = tid; i < MFEAT * DK; i += 256) S[i] = 0.f;
    for (int i = tid; i < DK * QPS;   i += 256) Sbf[i] = (bf16_t)0.f;
    Sk[tid] = 0.f;
    __syncthreads();

    const int t0 = wave * 2;                       // 2 of 16 output tiles/wave

    for (int c = 0; c < LL / CHUNK; ++c) {
        // ---- stage qp/kp chunk (vectorized), kp^T, and v^T ----
        {
            int j = tid >> 2, seg = (tid & 3) * 64;
            const uint4* gq = (const uint4*)(qpg + (long)(c * CHUNK + j) * (HEADS * MFEAT) + seg);
            const uint4* gk = (const uint4*)(kpg + (long)(c * CHUNK + j) * (HEADS * MFEAT) + seg);
            uint4* dq = (uint4*)&qpc[j * QPS + seg];
            uint4* dk = (uint4*)&kpc[j * QPS + seg];
            uint4 kbuf[8];
            #pragma unroll
            for (int u = 0; u < 8; ++u) {
                uint4 qv = gq[u]; uint4 kv = gk[u];
                dq[u] = qv; dk[u] = kv; kbuf[u] = kv;
            }
            const bf16_t* kb = (const bf16_t*)kbuf;     // scatter kp^T from regs
            #pragma unroll
            for (int mm = 0; mm < 64; ++mm) kpT[(seg + mm) * VTS + j] = kb[mm];

            int dseg = (tid & 3) * 16;
            const float* gv = vg + (long)(c * CHUNK + j) * DIMM + dseg;
            #pragma unroll
            for (int d = 0; d < 16; ++d) vT[(dseg + d) * VTS + j] = (bf16_t)gv[d];
        }
        __syncthreads();

        // ---- A = qp kp^T  (K = 256), mask causal, stash bf16 ----
        #pragma unroll
        for (int tt = 0; tt < 2; ++tt) {
            int t = t0 + tt, tm = t >> 2, tn = t & 3;
            v8f a = {};
            for (int ks = 0; ks < MFEAT; ks += 32) {
                v16bf af  = frag_a_lds(&qpc[(tm * 16) * QPS + ks], QPS);
                v16bf bfv = frag_b_lds(&kpc[(tn * 16) * QPS + ks], QPS);
                a = WMMA_BF16(af, bfv, a);
            }
            int n = lane & 15, hi = lane >> 4;
            int i0 = tm * 16 + hi * 8, jj = tn * 16 + n;
            #pragma unroll
            for (int r = 0; r < 8; ++r) {
                float val = (jj <= i0 + r) ? a[r] : 0.f;
                Am[(i0 + r) * VTS + jj] = (bf16_t)val;
            }
        }
        __syncthreads();

        // ---- den_i = qp_i . Sk_prev + rowsum(Am_i) ----
        if (tid < CHUNK) {
            float d = 0.f;
            for (int m = 0; m < MFEAT; ++m) d += (float)qpc[tid * QPS + m] * Sk[m];
            for (int j = 0; j <= tid; ++j)  d += (float)Am[tid * VTS + j];
            den[tid] = d;
        }
        __syncthreads();

        // ---- num = qp S_prev + Am v  (B-fragments from bf16 S^T shadow) ----
        v8f numacc[2];
        #pragma unroll
        for (int tt = 0; tt < 2; ++tt) {
            int t = t0 + tt, tm = t >> 2, tn = t & 3;
            v8f a = {};
            for (int ks = 0; ks < MFEAT; ks += 32) {
                v16bf af  = frag_a_lds(&qpc[(tm * 16) * QPS + ks], QPS);
                v16bf bfv = frag_b_lds(&Sbf[(tn * 16) * QPS + ks], QPS);
                a = WMMA_BF16(af, bfv, a);
            }
            for (int ks = 0; ks < CHUNK; ks += 32) {
                v16bf af  = frag_a_lds(&Am[(tm * 16) * VTS + ks], VTS);
                v16bf bfv = frag_b_lds(&vT[(tn * 16) * VTS + ks], VTS);
                a = WMMA_BF16(af, bfv, a);
            }
            numacc[tt] = a;
        }
        // ---- out = num / den ----
        {
            int n = lane & 15, hi = lane >> 4;
            #pragma unroll
            for (int tt = 0; tt < 2; ++tt) {
                int t = t0 + tt, tm = t >> 2, tn = t & 3;
                int i0 = tm * 16 + hi * 8;
                #pragma unroll
                for (int r = 0; r < 8; ++r) {
                    int i = i0 + r;
                    og[(long)(c * CHUNK + i) * DIMM + tn * 16 + n] = numacc[tt][r] / den[i];
                }
            }
        }
        __syncthreads();

        // ---- S += kp^T v  (A from staged kp^T, C in/out fp32 LDS),
        //      Sbf^T shadow refreshed from the same registers,
        //      Sk += colsum(kp) from contiguous kp^T rows ----
        #pragma unroll
        for (int mt = 0; mt < 2; ++mt) {
            int tm = wave * 2 + mt;                     // m-tile 0..15
            #pragma unroll
            for (int tn = 0; tn < 4; ++tn) {
                int n = lane & 15, hi = lane >> 4;
                v8f cacc;
                #pragma unroll
                for (int r = 0; r < 8; ++r)
                    cacc[r] = S[(tm * 16 + hi * 8 + r) * DK + tn * 16 + n];
                for (int ks = 0; ks < CHUNK; ks += 32) {
                    v16bf af  = frag_a_lds(&kpT[(tm * 16) * VTS + ks], VTS);
                    v16bf bfv = frag_b_lds(&vT[(tn * 16) * VTS + ks], VTS);
                    cacc = WMMA_BF16(af, bfv, cacc);
                }
                #pragma unroll
                for (int r = 0; r < 8; ++r) {
                    int m = tm * 16 + hi * 8 + r, d = tn * 16 + n;
                    S[m * DK + d]   = cacc[r];
                    Sbf[d * QPS + m] = (bf16_t)cacc[r];
                }
            }
        }
        {
            float s = Sk[tid];
            for (int j = 0; j < CHUNK; ++j) s += (float)kpT[tid * VTS + j];
            Sk[tid] = s;
        }
        __syncthreads();
    }
}

// ---------------------------------------------------------------------------
extern "C" void kernel_launch(void* const* d_in, const int* in_sizes, int n_in,
                              void* d_out, int out_size, void* d_ws, size_t ws_size,
                              hipStream_t stream) {
    const float* query = (const float*)d_in[0];
    const float* key_  = (const float*)d_in[1];
    const float* value = (const float*)d_in[2];
    const float* proj  = (const float*)d_in[3];
    const float* wq = (const float*)d_in[4];  const float* bq = (const float*)d_in[5];
    const float* wk = (const float*)d_in[6];  const float* bk = (const float*)d_in[7];
    const float* wv = (const float*)d_in[8];  const float* bv = (const float*)d_in[9];
    const float* wo = (const float*)d_in[10]; const float* bo = (const float*)d_in[11];
    float* out = (float*)d_out;

    char* ws = (char*)d_ws;
    float*  q  = (float*)(ws);                          // 16 MB
    float*  k  = (float*)(ws + ((size_t)16 << 20));     // 16 MB
    float*  v  = (float*)(ws + ((size_t)32 << 20));     // 16 MB
    bf16_t* qp = (bf16_t*)(ws + ((size_t)48 << 20));    // 32 MB
    bf16_t* kp = (bf16_t*)(ws + ((size_t)80 << 20));    // 32 MB
    float*  attn = q;                                   // q is dead after qp

    dim3 blk(256);
    dim3 gproj(DIMM / BN, NROWS / BM, 1);               // (8, 64)
    gemm_nt<<<gproj, blk, 0, stream>>>(query, DIMM, 0, wq, DIMM, bq, 1.f, 0,
                                       q, nullptr, DIMM, 0, DIMM);
    gemm_nt<<<gproj, blk, 0, stream>>>(key_,  DIMM, 0, wk, DIMM, bk, 1.f, 0,
                                       k, nullptr, DIMM, 0, DIMM);
    gemm_nt<<<gproj, blk, 0, stream>>>(value, DIMM, 0, wv, DIMM, bv, 1.f, 0,
                                       v, nullptr, DIMM, 0, DIMM);

    dim3 gfeat(MFEAT / BN, NROWS / BM, HEADS);          // (4, 64, 8)
    const float ratio = 0.0625f;                        // 1/sqrt(256)
    gemm_nt<<<gfeat, blk, 0, stream>>>(q, DIMM, DK, proj, DK, nullptr, ratio, 1,
                                       nullptr, qp, HEADS * MFEAT, MFEAT, DK);
    gemm_nt<<<gfeat, blk, 0, stream>>>(k, DIMM, DK, proj, DK, nullptr, ratio, 1,
                                       nullptr, kp, HEADS * MFEAT, MFEAT, DK);

    favor_scan<<<dim3(BB * HEADS), blk, SMEM_TOTAL, stream>>>(qp, kp, v, attn);

    gemm_nt<<<gproj, blk, 0, stream>>>(attn, DIMM, 0, wo, DIMM, bo, 1.f, 0,
                                       out, nullptr, DIMM, 0, DIMM);
}
